// BatchedGAT_89928025243997
// MI455X (gfx1250) — matrix operand order repaired and checked by reading, not compiled, not run on hardware
//
#include <hip/hip_runtime.h>

typedef __attribute__((ext_vector_type(16))) __bf16       v16bf;
typedef __attribute__((ext_vector_type(8)))  float        v8f;
typedef __attribute__((ext_vector_type(8)))  unsigned int v8u;

#define B_   16
#define N_   1024
#define F_   128
#define H_   4
#define DH_  64
#define OD_  256
#define NEG_SLOPE 0.2f
#define MASK_VAL  (-1e9f)

// Native f32->bf16 conversion (lowers to v_cvt_pk_bf16_f32 on gfx1250).
static __device__ __forceinline__ __bf16 f2bf(float f) {
  return (__bf16)f;
}
static __device__ __forceinline__ unsigned short f2bfu(float f) {
  return __builtin_bit_cast(unsigned short, (__bf16)f);
}

// ---------------------------------------------------------------------------
// Kernel 0: prepack W [128][256] f32 into per-lane WMMA-B fragment order, bf16.
// B-frag order used here (contiguous k, same as the A-frags): lane l -> column n = l&15, K = kb + e,
// kb = (l<16 ? 0 : 16), e = 0..15 consecutive -> 32 contiguous bytes per lane.
// Wpk[kt][ct][lane][e] , kt = K-tile of 32 (4), ct = col-tile of 16 (16).
// ---------------------------------------------------------------------------
__global__ void gat_prepackW(const float* __restrict__ W,
                             unsigned short* __restrict__ Wpk) {
  int kt = blockIdx.x;          // 0..3
  int ct = blockIdx.y;          // 0..15
  int l  = threadIdx.x;         // 0..31
  int col = ct * 16 + (l & 15);
  int kb  = (l < 16) ? 0 : 16;
  unsigned short* dst = Wpk + ((size_t)((kt * 16 + ct) * 32 + l)) * 16;
#pragma unroll
  for (int e = 0; e < 16; ++e) {
    int K = kt * 32 + kb + e;
    dst[e] = f2bfu(W[(size_t)K * OD_ + col]);
  }
}

// ---------------------------------------------------------------------------
// Kernel 1: projection h = x @ W via WMMA bf16, plus e_src/e_dst dot products,
// writing hT[b][head][d][n] in bf16 (transposed so phase-2 B-frags are
// contiguous). 1024 blocks x 128 threads; wave w handles head w (cols w*64..).
// ---------------------------------------------------------------------------
__global__ __launch_bounds__(128) void gat_proj(
    const float* __restrict__ x,
    const unsigned short* __restrict__ Wpk,
    const float* __restrict__ a_src, const float* __restrict__ a_dst,
    unsigned short* __restrict__ hT,
    float* __restrict__ esg, float* __restrict__ edg) {
  const int tid  = threadIdx.x;
  const int wv   = tid >> 5;
  const int lane = tid & 31;
  const int i0   = blockIdx.x * 16;       // flat row tile over B*N
  const int b    = i0 >> 10;
  const int n0   = i0 & (N_ - 1);
  const int mrow = lane & 15;
  const int hi   = lane >> 4;
  const int c0   = hi ? 16 : 0;           // A-frag K sub-base
  const int h    = wv;                    // head == wave

  v8f acc[4];
  v8f zv = {};
#pragma unroll
  for (int t = 0; t < 4; ++t) acc[t] = zv;

  const float* xrow = x + (size_t)(i0 + mrow) * F_;
#pragma unroll
  for (int kt = 0; kt < 4; ++kt) {
    const float* xr = xrow + kt * 32 + c0;
    float4 a0 = *(const float4*)(xr);
    float4 a1 = *(const float4*)(xr + 4);
    float4 a2 = *(const float4*)(xr + 8);
    float4 a3 = *(const float4*)(xr + 12);
    v16bf af;
    af[0]  = f2bf(a0.x); af[1]  = f2bf(a0.y); af[2]  = f2bf(a0.z); af[3]  = f2bf(a0.w);
    af[4]  = f2bf(a1.x); af[5]  = f2bf(a1.y); af[6]  = f2bf(a1.z); af[7]  = f2bf(a1.w);
    af[8]  = f2bf(a2.x); af[9]  = f2bf(a2.y); af[10] = f2bf(a2.z); af[11] = f2bf(a2.w);
    af[12] = f2bf(a3.x); af[13] = f2bf(a3.y); af[14] = f2bf(a3.z); af[15] = f2bf(a3.w);
#pragma unroll
    for (int tl = 0; tl < 4; ++tl) {
      int ct = h * 4 + tl;
      v16bf bf = __builtin_bit_cast(
          v16bf, *(const v8u*)(Wpk + ((size_t)((kt * 16 + ct) * 32 + lane)) * 16));
      acc[tl] = __builtin_amdgcn_wmma_f32_16x16x32_bf16(
          false, af, false, bf, (short)0, acc[tl], false, false);
    }
  }

  // --- e_src / e_dst: per-row dot(h_row_head, a_src/a_dst) via shuffles ---
  float asv[4], adv[4];
#pragma unroll
  for (int tl = 0; tl < 4; ++tl) {
    int d = tl * 16 + mrow;
    asv[tl] = a_src[h * DH_ + d];
    adv[tl] = a_dst[h * DH_ + d];
  }
#pragma unroll
  for (int r = 0; r < 8; ++r) {
    float ss = 0.f, dd = 0.f;
#pragma unroll
    for (int tl = 0; tl < 4; ++tl) {
      ss += acc[tl][r] * asv[tl];
      dd += acc[tl][r] * adv[tl];
    }
#pragma unroll
    for (int off = 1; off < 16; off <<= 1) {
      ss += __shfl_xor(ss, off, 32);
      dd += __shfl_xor(dd, off, 32);
    }
    if (mrow == 0) {
      int row = i0 + r + hi * 8;
      esg[(size_t)row * H_ + h] = ss;
      edg[(size_t)row * H_ + h] = dd;
    }
  }

  // --- write hT[b][head][d][n] bf16 (packed 16B stores) ---
#pragma unroll
  for (int tl = 0; tl < 4; ++tl) {
    int d = tl * 16 + mrow;
    unsigned short* hp =
        hT + (((size_t)(b * H_ + h)) * DH_ + d) * N_ + n0 + hi * 8;
    uint4 pk;
    pk.x = (unsigned)f2bfu(acc[tl][0]) | ((unsigned)f2bfu(acc[tl][1]) << 16);
    pk.y = (unsigned)f2bfu(acc[tl][2]) | ((unsigned)f2bfu(acc[tl][3]) << 16);
    pk.z = (unsigned)f2bfu(acc[tl][4]) | ((unsigned)f2bfu(acc[tl][5]) << 16);
    pk.w = (unsigned)f2bfu(acc[tl][6]) | ((unsigned)f2bfu(acc[tl][7]) << 16);
    *(uint4*)hp = pk;
  }
}

// ---------------------------------------------------------------------------
// Kernel 2: masked softmax over neighbors + WMMA aggregation.
// One block per (b, 16-row tile): adj read ONCE -> 1-bit mask in LDS.
// Wave w owns head w; alpha A-fragments are built directly in registers.
// B-fragments (hT, L2-resident) are double-buffered so the 8x b128 loads of
// tile j+1 are in flight while tile j's alpha-exp VALU and WMMAs execute.
// ---------------------------------------------------------------------------
__global__ __launch_bounds__(128) void gat_attn_agg(
    const float* __restrict__ adj,
    const unsigned short* __restrict__ hT,
    const float* __restrict__ esg, const float* __restrict__ edg,
    const float* __restrict__ bias,
    float* __restrict__ out) {
  __shared__ __align__(16) float es_l[H_ * N_];   // 16 KB: e_src per head
  __shared__ unsigned mask_l[16 * 32];            // 2 KB: adj>0.5 bitmask
  __shared__ float ed_l[16 * H_];
  __shared__ float mrow_l[16 * H_];
  __shared__ float sinv_l[16 * H_];

  const int tid  = threadIdx.x;
  const int wv   = tid >> 5;
  const int lane = tid & 31;
  const int bid  = blockIdx.x;
  const int b    = bid >> 6;
  const int n0   = (bid & 63) * 16;

  // --- stage e_src (coalesced), e_dst ---
  const float* esb = esg + (size_t)b * N_ * H_;
  for (int t = tid; t < N_ * H_; t += 128) {
    int j = t >> 2, h2 = t & 3;
    es_l[h2 * N_ + j] = esb[t];
  }
  if (tid < 64) ed_l[tid] = edg[((size_t)b * N_ + n0) * H_ + tid];

  // --- adj -> bitmask (adj read exactly once from HBM) ---
  const float* adjb = adj + ((size_t)b * N_ + n0) * N_;
#pragma unroll
  for (int rr = 0; rr < 4; ++rr) {
    int r = wv * 4 + rr;
    const float* arow = adjb + (size_t)r * N_;
    for (int c = 0; c < 32; ++c) {
      unsigned bal = __builtin_amdgcn_ballot_w32(arow[c * 32 + lane] > 0.5f);
      if (lane == 0) mask_l[r * 32 + c] = bal;
    }
  }
  __syncthreads();

  // --- pass 1: row max & exp-sum per (row, head) ---
  for (int p = wv * 16; p < wv * 16 + 16; ++p) {
    int r = p >> 2, h2 = p & 3;
    float edv = ed_l[r * 4 + h2];
    const float* esh = es_l + h2 * N_;
    const unsigned* mk = mask_l + r * 32;
    float m = -3.0e38f;
    for (int c = 0; c < 32; ++c) {
      float e = edv + esh[c * 32 + lane];
      e = e > 0.f ? e : NEG_SLOPE * e;
      e = ((mk[c] >> lane) & 1u) ? e : MASK_VAL;
      m = fmaxf(m, e);
    }
#pragma unroll
    for (int off = 16; off >= 1; off >>= 1) m = fmaxf(m, __shfl_xor(m, off, 32));
    float s = 0.f;
    for (int c = 0; c < 32; ++c) {
      float e = edv + esh[c * 32 + lane];
      e = e > 0.f ? e : NEG_SLOPE * e;
      e = ((mk[c] >> lane) & 1u) ? e : MASK_VAL;
      s += __expf(e - m);
    }
#pragma unroll
    for (int off = 16; off >= 1; off >>= 1) s += __shfl_xor(s, off, 32);
    if (lane == 0) { mrow_l[p] = m; sinv_l[p] = 1.0f / s; }
  }
  __syncthreads();

  // --- pass 2: alpha A-frags in registers, double-buffered B, WMMA ---
  const int h    = wv;            // head == wave
  const int mrow = lane & 15;     // A row; B/C column
  const int hi   = lane >> 4;
  const int c0   = hi ? 16 : 0;   // A-frag K sub-base
  const int kb   = hi ? 16 : 0;   // B-frag K base
  const float edv = ed_l[mrow * 4 + h];
  const float mv  = mrow_l[mrow * 4 + h];
  const float sv  = sinv_l[mrow * 4 + h];
  const unsigned short* hTh = hT + ((size_t)(b * H_ + h)) * DH_ * N_;
  const unsigned* mk = mask_l + mrow * 32;
  const float4* esh4 = (const float4*)(es_l + h * N_);

  v8f acc[4];
  v8f zv = {};
#pragma unroll
  for (int t = 0; t < 4; ++t) acc[t] = zv;

  // prologue: load B fragments of tile j0 = 0
  v8u braw[4];
#pragma unroll
  for (int t = 0; t < 4; ++t)
    braw[t] = *(const v8u*)(hTh + ((size_t)(t * 16 + mrow)) * N_ + kb);

  for (int j0 = 0; j0 < N_; j0 += 32) {
    // issue next tile's B loads early (in flight under A-build + WMMAs)
    v8u bnext[4];
    const int jn = j0 + 32;
    if (jn < N_) {
#pragma unroll
      for (int t = 0; t < 4; ++t)
        bnext[t] = *(const v8u*)(hTh + ((size_t)(t * 16 + mrow)) * N_ + jn + kb);
    }

    // build alpha A-fragment for tile j0 (registers only)
    unsigned mw = mk[j0 >> 5];
    int b4 = (j0 >> 2) + (c0 >> 2);
    float4 r0 = esh4[b4];
    float4 r1 = esh4[b4 + 1];
    float4 r2 = esh4[b4 + 2];
    float4 r3 = esh4[b4 + 3];
    float ev[16] = {r0.x, r0.y, r0.z, r0.w, r1.x, r1.y, r1.z, r1.w,
                    r2.x, r2.y, r2.z, r2.w, r3.x, r3.y, r3.z, r3.w};
    v16bf af;
#pragma unroll
    for (int k = 0; k < 16; ++k) {
      int K = c0 + k;
      float e = edv + ev[k];
      e = e > 0.f ? e : NEG_SLOPE * e;
      e = ((mw >> K) & 1u) ? e : MASK_VAL;
      af[k] = f2bf(__expf(e - mv) * sv);
    }

#pragma unroll
    for (int t = 0; t < 4; ++t) {
      acc[t] = __builtin_amdgcn_wmma_f32_16x16x32_bf16(
          false, af, false, __builtin_bit_cast(v16bf, braw[t]),
          (short)0, acc[t], false, false);
    }
#pragma unroll
    for (int t = 0; t < 4; ++t) braw[t] = bnext[t];
  }

  // --- store out[b][n][head*64 + d] + bias ---
#pragma unroll
  for (int t = 0; t < 4; ++t) {
    int col = h * DH_ + t * 16 + mrow;
    float bv = bias[col];
#pragma unroll
    for (int r = 0; r < 8; ++r) {
      size_t row = (size_t)b * N_ + n0 + r + hi * 8;
      out[row * OD_ + col] = acc[t][r] + bv;
    }
  }
}

// ---------------------------------------------------------------------------
extern "C" void kernel_launch(void* const* d_in, const int* in_sizes, int n_in,
                              void* d_out, int out_size, void* d_ws, size_t ws_size,
                              hipStream_t stream) {
  const float* x     = (const float*)d_in[0];
  const float* adj   = (const float*)d_in[1];
  const float* W     = (const float*)d_in[2];
  const float* a_src = (const float*)d_in[3];
  const float* a_dst = (const float*)d_in[4];
  const float* bias  = (const float*)d_in[5];
  float* out = (float*)d_out;

  char* ws = (char*)d_ws;
  // layout: hT bf16 [B][H][DH][N] = 8 MB | esg 256KB | edg 256KB | Wpk 64KB
  unsigned short* hT  = (unsigned short*)ws;
  float*          esg = (float*)(ws + (size_t)8 * 1024 * 1024);
  float*          edg = (float*)(ws + (size_t)8 * 1024 * 1024 + 256 * 1024);
  unsigned short* Wpk = (unsigned short*)(ws + (size_t)8 * 1024 * 1024 + 512 * 1024);

  gat_prepackW<<<dim3(4, 16), 32, 0, stream>>>(W, Wpk);
  gat_proj<<<(B_ * N_) / 16, 128, 0, stream>>>(x, Wpk, a_src, a_dst, hT, esg, edg);
  gat_attn_agg<<<(B_ * N_) / 16, 128, 0, stream>>>(adj, hT, esg, edg, bias, out);
}
